// GCN_30279519437683
// MI455X (gfx1250) — matrix operand order repaired and checked by reading, not compiled, not run on hardware
//
#include <hip/hip_runtime.h>
#include <hip/hip_fp16.h>

// ---- problem constants (from reference) ----
#define NNODES 100000   // = 6250 * 16 -> M tiles exactly
#define NEDGES 1600000
#define IN_F   128
#define HID    64
#define NCLS   16

typedef __attribute__((ext_vector_type(16))) _Float16 v16h;
typedef __attribute__((ext_vector_type(8)))  _Float16 v8h;
typedef __attribute__((ext_vector_type(8)))  float    v8f;

__device__ __forceinline__ void atomic_add_f32(float* p, float v) {
    // relaxed, device scope -> hw global_atomic_add_f32 on gfx1250
    __hip_atomic_fetch_add(p, v, __ATOMIC_RELAXED, __HIP_MEMORY_SCOPE_AGENT);
}

// ---------------- zero fill ----------------
__global__ void k_zero(float* __restrict__ p, long long n) {
    long long i = (long long)blockIdx.x * blockDim.x + threadIdx.x;
    long long stride = (long long)gridDim.x * blockDim.x;
    for (; i < n; i += stride) p[i] = 0.0f;
}

// ---------------- degree count ----------------
__global__ void k_degrees(const int* __restrict__ src, const int* __restrict__ dst,
                          float* __restrict__ degO, float* __restrict__ degI, int E) {
    int e = blockIdx.x * blockDim.x + threadIdx.x;
    if (e < E) {
        atomic_add_f32(&degO[src[e]], 1.0f);
        atomic_add_f32(&degI[dst[e]], 1.0f);
    }
}

// deg -> rsqrt(clip(deg,1)) in place (one launch spans rsO, a 32-float pad and rsI[0..99967])
__global__ void k_rsqrt(float* __restrict__ d, int n) {
    int i = blockIdx.x * blockDim.x + threadIdx.x;
    if (i < n) d[i] = rsqrtf(fmaxf(d[i], 1.0f));
}

// ---------------- f32 -> f16 convert (packed 2/thread) ----------------
__global__ void k_cvt_f16(const float* __restrict__ x, _Float16* __restrict__ y, long long n2) {
    long long i = (long long)blockIdx.x * blockDim.x + threadIdx.x;   // pair index
    if (i >= n2) return;
    float2 v = *(const float2*)(x + 2 * i);
    _Float16 h0 = (_Float16)v.x, h1 = (_Float16)v.y;
    *(__attribute__((ext_vector_type(2))) _Float16*)(y + 2 * i) =
        (__attribute__((ext_vector_type(2))) _Float16){h0, h1};
}

// ---------------- W[K][Nd] f32  ->  Wt[Nd][K] f16 ----------------
__global__ void k_wt(const float* __restrict__ W, _Float16* __restrict__ Wt, int K, int Nd) {
    int idx = blockIdx.x * blockDim.x + threadIdx.x;
    if (idx >= K * Nd) return;
    int k = idx / Nd, n = idx % Nd;
    Wt[(size_t)n * K + k] = (_Float16)W[idx];
}

// ---------------- WMMA GEMM ----------------
// Y[m][n] = act( rowScale[m] * (A[M,K]f16 @ W[K,Nd]) + bias[n] )
// A row-major f16; Wt = W transposed, f16 [Nd][K].
// One wave per 16x16 C tile, 2 waves/block along M, fully unrolled K loop.
// Fragment layouts per CDNA5 ISA 7.12.2:
//   A (16x32 f16): lane holds row M=lane%16; halves = K {hiA..hiA+7, 16+hiA..+7}, hiA=8*(lane>=16)
//   B (32x16 f16): lane holds col N=lane%16; halves = K {hiB..hiB+7, 16+hiB..+7}, hiB=8*(lane>=16)
//   C (16x16 f32): VGPR r -> (M = r + 8*(lane>=16), N = lane%16)
template <int K, int ND, bool RELU, bool F16OUT>
__global__ __launch_bounds__(64) void k_gemm(
    const _Float16* __restrict__ A, const _Float16* __restrict__ Wt,
    const float* __restrict__ rowScale, const float* __restrict__ bias,
    float* __restrict__ Yf, _Float16* __restrict__ Yh)
{
    const int wave = threadIdx.x >> 5;
    const int lane = threadIdx.x & 31;
    const int mt   = blockIdx.x * 2 + wave;      // grid.x = NNODES/32
    const int nt   = blockIdx.y;
    const int mrow = mt * 16 + (lane & 15);
    const int ncol = nt * 16 + (lane & 15);
    const int hiA  = (lane >> 4) * 8;
    const int hiB  = (lane >> 4) * 8;

    const _Float16* ar = A  + (size_t)mrow * K;
    const _Float16* br = Wt + (size_t)ncol * K;

    v8f acc = {};
    #pragma unroll
    for (int k0 = 0; k0 < K; k0 += 32) {
        union { v16h v; v8h h[2]; } a, b;
        a.h[0] = *(const v8h*)(ar + k0 + hiA);        // global_load_b128
        a.h[1] = *(const v8h*)(ar + k0 + 16 + hiA);   // global_load_b128
        b.h[0] = *(const v8h*)(br + k0 + hiB);        // global_load_b128
        b.h[1] = *(const v8h*)(br + k0 + 16 + hiB);   // global_load_b128
        acc = __builtin_amdgcn_wmma_f32_16x16x32_f16(
            false, a.v, false, b.v, (short)0, acc, false, false);
    }

    const int mBase = mt * 16 + (lane >> 4) * 8;
    const float bv  = bias ? bias[ncol] : 0.0f;
    #pragma unroll
    for (int r = 0; r < 8; ++r) {
        int m = mBase + r;
        float v = acc[r] * rowScale[m] + bv;
        if (RELU) v = fmaxf(v, 0.0f);
        if (F16OUT) Yh[(size_t)m * ND + ncol] = (_Float16)v;
        else        Yf[(size_t)m * ND + ncol] = v;
    }
}

// ---------------- edge scatter-add: AGG[dst] += H[src] * srcScale[src] ----------
// F/4 threads per edge, float4 gather + 4 hw f32 atomics.
__global__ void k_scatter(const int* __restrict__ src, const int* __restrict__ dst,
                          const float* __restrict__ H,
                          const float* __restrict__ srcScale,   // nullable
                          float* __restrict__ AGG, int E, int F, int qshift) {
    long long t = (long long)blockIdx.x * blockDim.x + threadIdx.x;
    long long total = (long long)E << qshift;      // E * (F/4)
    if (t >= total) return;
    int e = (int)(t >> qshift);
    int c = (int)(t & ((1 << qshift) - 1)) << 2;
    int s = src[e], d = dst[e];
    float4 h4 = *(const float4*)(H + (size_t)s * F + c);
    float sc = srcScale ? srcScale[s] : 1.0f;
    float* base = AGG + (size_t)d * F + c;
    atomic_add_f32(base + 0, h4.x * sc);
    atomic_add_f32(base + 1, h4.y * sc);
    atomic_add_f32(base + 2, h4.z * sc);
    atomic_add_f32(base + 3, h4.w * sc);
}

// ---------------- per-node finalize: Y = act(AGG * rin[row] + b[col]) ----------
__global__ void k_finalize(const float* __restrict__ AGG, const float* __restrict__ rin,
                           const float* __restrict__ b, float* __restrict__ Y,
                           long long total, int fshift, int relu) {
    long long i = (long long)blockIdx.x * blockDim.x + threadIdx.x;
    if (i >= total) return;
    int row = (int)(i >> fshift);
    int col = (int)(i & ((1 << fshift) - 1));
    float v = AGG[i] * rin[row] + b[col];
    if (relu) v = fmaxf(v, 0.0f);
    Y[i] = v;
}

extern "C" void kernel_launch(void* const* d_in, const int* in_sizes, int n_in,
                              void* d_out, int out_size, void* d_ws, size_t ws_size,
                              hipStream_t stream) {
    const float* feats = (const float*)d_in[0];
    const int*   src   = (const int*)  d_in[1];
    const int*   dst   = (const int*)  d_in[2];
    const float* W0    = (const float*)d_in[3];
    const float* b0    = (const float*)d_in[4];
    const float* W1    = (const float*)d_in[5];
    const float* b1    = (const float*)d_in[6];
    const float* W2    = (const float*)d_in[7];
    const float* b2    = (const float*)d_in[8];
    float* out = (float*)d_out;

    // ---- workspace layout (256B aligned chunks) ----
    char* p = (char*)d_ws;
    auto take = [&](size_t bytes) { char* q = p; p += (bytes + 255) & ~(size_t)255; return q; };
    float*    rsO  = (float*)   take(NNODES * 4);                    // rsqrt(deg_out)
    float*    rsI  = (float*)   take(NNODES * 4);                    // rsqrt(deg_in)
    float*    bufA = (float*)   take((size_t)NNODES * HID * 4);      // f32 scratch
    float*    bufB = (float*)   take((size_t)NNODES * HID * 4);      // f32 scratch
    _Float16* XH   = (_Float16*)take((size_t)NNODES * IN_F * 2);     // feats f16 / reuse
    _Float16* Wt0  = (_Float16*)take(IN_F * HID * 2);
    _Float16* Wt1  = (_Float16*)take(HID * HID * 2);
    _Float16* Wt2  = (_Float16*)take(HID * NCLS * 2);
    if ((size_t)(p - (char*)d_ws) > ws_size) return;
    _Float16* AGG1h = XH;                         // NNODES*HID halves (reuse after L0)
    _Float16* X2h   = XH + (size_t)NNODES * HID;  // NNODES*HID halves

    const long long nBig = (long long)NNODES * HID;     // 6.4M
    const long long nOut = (long long)NNODES * NCLS;    // 1.6M

    // ---- 0) prep: degrees + f16 operands ----
    k_zero<<<512, 256, 0, stream>>>(rsO, 2LL * NNODES);
    k_degrees<<<(NEDGES + 255) / 256, 256, 0, stream>>>(src, dst, rsO, rsI, NEDGES);
    k_rsqrt<<<(2 * NNODES + 255) / 256, 256, 0, stream>>>(rsO, 2 * NNODES);
    k_cvt_f16<<<(int)(((long long)NNODES * IN_F / 2 + 255) / 256), 256, 0, stream>>>(
        feats, XH, (long long)NNODES * IN_F / 2);
    k_wt<<<(IN_F * HID + 255) / 256, 256, 0, stream>>>(W0, Wt0, IN_F, HID);
    k_wt<<<(HID * HID + 255) / 256, 256, 0, stream>>>(W1, Wt1, HID, HID);
    k_wt<<<(HID * NCLS + 255) / 256, 256, 0, stream>>>(W2, Wt2, HID, NCLS);

    dim3 blk(64);
    dim3 grid64(NNODES / 32, HID / 16);   // 3125 x 4
    dim3 grid16(NNODES / 32, NCLS / 16);  // 3125 x 1

    // ---- 1) Layer 0 (pre-mult): H0 = diag(rsO) * (X @ W0) -> bufA (f32) ----
    k_gemm<IN_F, HID, false, false><<<grid64, blk, 0, stream>>>(XH, Wt0, rsO, nullptr, bufA, nullptr);
    k_zero<<<4096, 256, 0, stream>>>(bufB, nBig);
    k_scatter<<<(int)(((long long)NEDGES * 16 + 255) / 256), 256, 0, stream>>>(
        src, dst, bufA, nullptr, bufB, NEDGES, HID, 4);
    //    X1 = relu(AGG0 * rsI + b0) -> bufA  (XH now dead, safe to reuse)
    k_finalize<<<(int)((nBig + 255) / 256), 256, 0, stream>>>(bufB, rsI, b0, bufA, nBig, 6, 1);

    // ---- 2) Layer 1 (post-mult): AGG1 = scatter(X1 * rsO) -> bufB ----
    k_zero<<<4096, 256, 0, stream>>>(bufB, nBig);
    k_scatter<<<(int)(((long long)NEDGES * 16 + 255) / 256), 256, 0, stream>>>(
        src, dst, bufA, rsO, bufB, NEDGES, HID, 4);
    k_cvt_f16<<<(int)((nBig / 2 + 255) / 256), 256, 0, stream>>>(bufB, AGG1h, nBig / 2);
    //    X2 = relu( diag(rsI) * (AGG1 @ W1) + b1 ) -> X2h (f16, feeds L2 GEMM directly)
    k_gemm<HID, HID, true, true><<<grid64, blk, 0, stream>>>(AGG1h, Wt1, rsI, b1, nullptr, X2h);

    // ---- 3) Layer 2 (pre-mult): H2 = diag(rsO) * (X2 @ W2) -> bufA (f32, 1.6M) ----
    k_gemm<HID, NCLS, false, false><<<grid16, blk, 0, stream>>>(X2h, Wt2, rsO, nullptr, bufA, nullptr);
    k_zero<<<2048, 256, 0, stream>>>(out, nOut);
    k_scatter<<<(int)(((long long)NEDGES * 4 + 255) / 256), 256, 0, stream>>>(
        src, dst, bufA, nullptr, out, NEDGES, NCLS, 2);
    k_finalize<<<(int)((nOut + 255) / 256), 256, 0, stream>>>(out, rsI, b2, out, nOut, 4, 0);
}
